// Transformer_37821482009329
// MI455X (gfx1250) — matrix operand order repaired and checked by reading, not compiled, not run on hardware
//
#include <hip/hip_runtime.h>
#include <hip/hip_bf16.h>
#include <math.h>

#define DIM    512
#define DEPTH  4
#define VOCAB  32000
#define HEADS  512
#define FFDIM  2048
#define BATCH  2
#define SEQ    1024
#define ROWS   (BATCH * SEQ)        // 2048 token rows

#define CHUNKS 32                   // scan chunks per sequence
#define CLEN   (SEQ / CHUNKS)       // 32 steps per chunk

typedef __attribute__((ext_vector_type(16))) _Float16 v16h;
typedef __attribute__((ext_vector_type(8)))  float    v8f;

// ---------------------------------------------------------------------------
// Embedding gather: x[row, :] = emb[tokens[row], :]
// ---------------------------------------------------------------------------
__global__ __launch_bounds__(256) void embed_kernel(const int* __restrict__ tokens,
                                                    const float* __restrict__ emb,
                                                    float* __restrict__ x) {
    const int row = blockIdx.x;
    const int t   = tokens[row];
    const float* e = emb + (size_t)t * DIM;
    float* o = x + (size_t)row * DIM;
    o[threadIdx.x]       = e[threadIdx.x];
    o[threadIdx.x + 256] = e[threadIdx.x + 256];
}

// ---------------------------------------------------------------------------
// RMS-norm (F.normalize flavor): out = x / max(||x||,1e-12) * sqrt(DIM) * gamma
// ---------------------------------------------------------------------------
__global__ __launch_bounds__(256) void rmsnorm_kernel(const float* __restrict__ x,
                                                      const float* __restrict__ gamma,
                                                      float* __restrict__ out) {
    __shared__ float red[256];
    const int row = blockIdx.x;
    const float* xr = x + (size_t)row * DIM;
    const float v0 = xr[threadIdx.x];
    const float v1 = xr[threadIdx.x + 256];
    red[threadIdx.x] = v0 * v0 + v1 * v1;
    __syncthreads();
    #pragma unroll
    for (int s = 128; s > 0; s >>= 1) {
        if (threadIdx.x < s) red[threadIdx.x] += red[threadIdx.x + s];
        __syncthreads();
    }
    const float nrm   = sqrtf(red[0]);
    const float scale = 22.62741699796952f / fmaxf(nrm, 1e-12f);   // sqrt(512)
    float* o = out + (size_t)row * DIM;
    o[threadIdx.x]       = v0 * scale * gamma[threadIdx.x];
    o[threadIdx.x + 256] = v1 * scale * gamma[threadIdx.x + 256];
}

// ---------------------------------------------------------------------------
// GateLoop scan, chunked 3-pass decomposition.
//   gate g = sigmoid(|a|) * e^{i*angle(a)} ;  kv <- g*kv + k*v ;  out = q*Re(kv)
// The recurrence composes as (G, S) pairs: applying a chunk to state kv0 gives
// (prod g)*kv0 + S_local.  Pass1: per-chunk (G,S).  Pass2: serial exclusive
// carry over 32 chunks.  Pass3: replay chunk from its carry, emit output.
// Critical path: 32+32+32 steps on 32768 threads (vs 1024 steps on 1024).
// ---------------------------------------------------------------------------
__device__ __forceinline__ void gate_of(float re, float im, float& gr, float& gi) {
    const float mag = sqrtf(re * re + im * im);
    if (mag > 0.f) {
        const float s = (1.f / (1.f + expf(-mag))) / mag;
        gr = s * re; gi = s * im;
    } else {                       // angle(0)=0 -> gate = sigmoid(0) = 0.5
        gr = 0.5f; gi = 0.f;
    }
}

__global__ __launch_bounds__(256) void scan_part1(const float* __restrict__ qkv,
                                                  const float* __restrict__ alin,
                                                  float* __restrict__ gs) {
    const int t = blockIdx.x * 256 + threadIdx.x;     // h | b | c packed
    const int h = t & (HEADS - 1);
    const int b = (t >> 9) & (BATCH - 1);
    const int c = t >> 10;
    const float* qb = qkv  + (size_t)b * SEQ * (3 * DIM);
    const float* ab = alin + (size_t)b * SEQ * (2 * HEADS);
    float Gr = 1.f, Gi = 0.f, Sr = 0.f, Si = 0.f;
    const int nbeg = c * CLEN;
    for (int i = 0; i < CLEN; ++i) {
        const int n = nbeg + i;
        float gr, gi;
        gate_of(ab[(size_t)n * (2 * HEADS) + 2 * h],
                ab[(size_t)n * (2 * HEADS) + 2 * h + 1], gr, gi);
        const float* qr = qb + (size_t)n * (3 * DIM);
        const float kvv = qr[DIM + h] * qr[2 * DIM + h];
        const float sr = gr * Sr - gi * Si + kvv;
        const float si = gr * Si + gi * Sr;
        Sr = sr; Si = si;
        const float ggr = gr * Gr - gi * Gi;
        const float ggi = gr * Gi + gi * Gr;
        Gr = ggr; Gi = ggi;
    }
    float* o = gs + (size_t)((b * HEADS + h) * CHUNKS + c) * 4;
    o[0] = Gr; o[1] = Gi; o[2] = Sr; o[3] = Si;
}

__global__ __launch_bounds__(256) void scan_part2(const float* __restrict__ gs,
                                                  float* __restrict__ cin) {
    const int t = blockIdx.x * 256 + threadIdx.x;     // 0 .. B*HEADS-1
    const float* g = gs + (size_t)t * CHUNKS * 4;
    float* o = cin + (size_t)t * CHUNKS * 2;
    float cr = 0.f, ci = 0.f;
    for (int c = 0; c < CHUNKS; ++c) {
        o[2 * c] = cr; o[2 * c + 1] = ci;             // exclusive carry
        const float Gr = g[4 * c], Gi = g[4 * c + 1];
        const float Sr = g[4 * c + 2], Si = g[4 * c + 3];
        const float nr = Gr * cr - Gi * ci + Sr;
        const float ni = Gr * ci + Gi * cr + Si;
        cr = nr; ci = ni;
    }
}

__global__ __launch_bounds__(256) void scan_part3(const float* __restrict__ qkv,
                                                  const float* __restrict__ alin,
                                                  const float* __restrict__ cin,
                                                  float* __restrict__ attn) {
    const int t = blockIdx.x * 256 + threadIdx.x;
    const int h = t & (HEADS - 1);
    const int b = (t >> 9) & (BATCH - 1);
    const int c = t >> 10;
    const float* qb = qkv  + (size_t)b * SEQ * (3 * DIM);
    const float* ab = alin + (size_t)b * SEQ * (2 * HEADS);
    float* ob       = attn + (size_t)b * SEQ * DIM;
    const float* ci0 = cin + (size_t)((b * HEADS + h) * CHUNKS + c) * 2;
    float kvr = ci0[0], kvi = ci0[1];
    const int nbeg = c * CLEN;
    for (int i = 0; i < CLEN; ++i) {
        const int n = nbeg + i;
        float gr, gi;
        gate_of(ab[(size_t)n * (2 * HEADS) + 2 * h],
                ab[(size_t)n * (2 * HEADS) + 2 * h + 1], gr, gi);
        const float* qr = qb + (size_t)n * (3 * DIM);
        const float q = qr[h], kvv = qr[DIM + h] * qr[2 * DIM + h];
        const float nr = gr * kvr - gi * kvi + kvv;
        const float ni = gr * kvi + gi * kvr;
        kvr = nr; kvi = ni;
        ob[(size_t)n * DIM + h] = q * kvr;
    }
}

// ---------------------------------------------------------------------------
// WMMA f16 GEMM with fused epilogue.  C[M,N] = epi(A[M,K] @ B[K,N]).
//   EPI 0: none                       EPI 1: + bias[col]
//   EPI 2: gelu_exact(+bias)          EPI 3: silu(acc) * extra[row,col]
//   EPI 4: + bias + extra (residual)  EPI 5: + extra (residual)
// Block tile 64x64, K-step 32; 8 wave32s; each wave = 16x32 via two
// v_wmma_f32_16x16x32_f16 sharing one A fragment.  M%64==N%64==K%32==0.
// ---------------------------------------------------------------------------
#define LDSSTRIDE 36   // halves; pads the 32-wide K dim against bank conflicts

template <int EPI>
__global__ __launch_bounds__(256) void wmma_gemm_kernel(
    const float* __restrict__ A, const float* __restrict__ Bw,
    float* __restrict__ C, const float* __restrict__ bias,
    const float* __restrict__ extra, int M, int N, int K) {

    __shared__ _Float16 As[64 * LDSSTRIDE];   // [m][k] f16
    __shared__ _Float16 Bs[64 * LDSSTRIDE];   // [n][k] f16 (transposed on store)

    const int tid  = threadIdx.x;
    const int lane = tid & 31;
    const int wid  = tid >> 5;
    const int wm   = wid & 3;        // 4 wave rows  (16 each)
    const int wn   = wid >> 2;       // 2 wave cols  (32 each)
    const int g    = lane >> 4;      // lane group 0/1
    const int ml   = lane & 15;
    const int m0   = blockIdx.y * 64;
    const int n0   = blockIdx.x * 64;

    const int ar = tid >> 2, ac = (tid & 3) * 8;   // A stage: 64 rows x 32 k
    const int br = tid >> 3, bc = (tid & 7) * 8;   // B stage: 32 k   x 64 cols

    v8f acc0 = {}; v8f acc1 = {};

    for (int k0 = 0; k0 < K; k0 += 32) {
        const float* ap = A  + (size_t)(m0 + ar) * K + k0 + ac;
        const float* bp = Bw + (size_t)(k0 + br) * N + n0 + bc;
        __builtin_prefetch(ap + 32, 0, 1);            // next K-tile
        __builtin_prefetch(bp + (size_t)32 * N, 0, 1);
        float areg[8], breg[8];
        #pragma unroll
        for (int j = 0; j < 8; ++j) areg[j] = ap[j];
        #pragma unroll
        for (int j = 0; j < 8; ++j) breg[j] = bp[j];

        __syncthreads();   // previous iteration's fragment reads done
        #pragma unroll
        for (int j = 0; j < 8; ++j) As[ar * LDSSTRIDE + ac + j] = (_Float16)areg[j];
        #pragma unroll
        for (int j = 0; j < 8; ++j) Bs[(bc + j) * LDSSTRIDE + br] = (_Float16)breg[j];
        __syncthreads();

        // A fragment 16x32: lane m = wm*16+ml; halves e -> K = e + g*16
        v16h af;
        const _Float16* arow = &As[(wm * 16 + ml) * LDSSTRIDE + g * 16];
        #pragma unroll
        for (int e = 0; e < 16; ++e) { af[e] = arow[e]; }

        // B fragments 32x16: lane n; halves e -> K = e + g*16 (contiguous in LDS)
        v16h bf0, bf1;
        const _Float16* b0 = &Bs[(wn * 32 + ml) * LDSSTRIDE + g * 16];
        const _Float16* b1 = &Bs[(wn * 32 + 16 + ml) * LDSSTRIDE + g * 16];
        #pragma unroll
        for (int e = 0; e < 16; ++e) { bf0[e] = b0[e]; bf1[e] = b1[e]; }

        acc0 = __builtin_amdgcn_wmma_f32_16x16x32_f16(false, af, false, bf0,
                                                      (short)0, acc0, false, false);
        acc1 = __builtin_amdgcn_wmma_f32_16x16x32_f16(false, af, false, bf1,
                                                      (short)0, acc1, false, false);
    }

    // Epilogue. C/D layout: VGPR r -> M = r + g*8, N = ml.
    #pragma unroll
    for (int r = 0; r < 8; ++r) {
        const int row = m0 + wm * 16 + g * 8 + r;
        #pragma unroll
        for (int s = 0; s < 2; ++s) {
            const int col = n0 + wn * 32 + s * 16 + ml;
            float val = (s == 0) ? acc0[r] : acc1[r];
            const size_t idx = (size_t)row * N + col;
            if (EPI == 1 || EPI == 2 || EPI == 4) val += bias[col];
            if (EPI == 2) val = 0.5f * val * (1.f + erff(val * 0.7071067811865475f));
            if (EPI == 3) { const float sv = val / (1.f + expf(-val)); val = sv * extra[idx]; }
            if (EPI == 4 || EPI == 5) val += extra[idx];
            C[idx] = val;
        }
    }
}

// ---------------------------------------------------------------------------
// Host orchestration
// ---------------------------------------------------------------------------
extern "C" void kernel_launch(void* const* d_in, const int* in_sizes, int n_in,
                              void* d_out, int out_size, void* d_ws, size_t ws_size,
                              hipStream_t stream) {
    const int*   tokens   = (const int*)  d_in[0];
    const float* emb      = (const float*)d_in[1];
    const float* g_attn   = (const float*)d_in[2];
    const float* Wqkv     = (const float*)d_in[3];
    const float* Wa       = (const float*)d_in[4];
    const float* ba       = (const float*)d_in[5];
    const float* Wg       = (const float*)d_in[6];
    const float* Wo       = (const float*)d_in[7];
    const float* g_ff     = (const float*)d_in[8];
    const float* W1       = (const float*)d_in[9];
    const float* b1       = (const float*)d_in[10];
    const float* W2       = (const float*)d_in[11];
    const float* b2       = (const float*)d_in[12];
    const float* g_final  = (const float*)d_in[13];
    const float* Wl       = (const float*)d_in[14];
    float* out = (float*)d_out;

    float* ws    = (float*)d_ws;                    // ~60 MB of f32 scratch
    float* x     = ws;                              // [2048,512]
    float* xn    = x    + (size_t)ROWS * DIM;       // [2048,512]
    float* qkv   = xn   + (size_t)ROWS * DIM;       // [2048,1536]
    float* alin  = qkv  + (size_t)ROWS * 3 * DIM;   // [2048,1024]
    float* attn  = alin + (size_t)ROWS * 2 * HEADS; // [2048,512]
    float* gbuf  = attn + (size_t)ROWS * DIM;       // [2048,512]
    float* ffh   = gbuf + (size_t)ROWS * DIM;       // [2048,2048]
    float* gsbuf = ffh  + (size_t)ROWS * FFDIM;     // [B*H, CHUNKS, 4]
    float* cinbf = gsbuf + (size_t)BATCH * HEADS * CHUNKS * 4; // [B*H, CHUNKS, 2]

    embed_kernel<<<ROWS, 256, 0, stream>>>(tokens, emb, x);

    for (int l = 0; l < DEPTH; ++l) {
        const float* lWqkv = Wqkv + (size_t)l * DIM * 3 * DIM;
        const float* lWa   = Wa   + (size_t)l * DIM * 2 * HEADS;
        const float* lba   = ba   + (size_t)l * 2 * HEADS;
        const float* lWg   = Wg   + (size_t)l * DIM * DIM;
        const float* lWo   = Wo   + (size_t)l * DIM * DIM;
        const float* lW1   = W1   + (size_t)l * DIM * FFDIM;
        const float* lb1   = b1   + (size_t)l * FFDIM;
        const float* lW2   = W2   + (size_t)l * FFDIM * DIM;
        const float* lb2   = b2   + (size_t)l * DIM;

        // --- attention block ---
        rmsnorm_kernel<<<ROWS, 256, 0, stream>>>(x, g_attn + (size_t)l * DIM, xn);
        wmma_gemm_kernel<0><<<dim3(3 * DIM / 64, ROWS / 64), 256, 0, stream>>>(
            xn, lWqkv, qkv, nullptr, nullptr, ROWS, 3 * DIM, DIM);
        wmma_gemm_kernel<1><<<dim3(2 * HEADS / 64, ROWS / 64), 256, 0, stream>>>(
            xn, lWa, alin, lba, nullptr, ROWS, 2 * HEADS, DIM);

        scan_part1<<<(BATCH * HEADS * CHUNKS) / 256, 256, 0, stream>>>(qkv, alin, gsbuf);
        scan_part2<<<(BATCH * HEADS) / 256, 256, 0, stream>>>(gsbuf, cinbf);
        scan_part3<<<(BATCH * HEADS * CHUNKS) / 256, 256, 0, stream>>>(qkv, alin, cinbf, attn);

        // gbuf = silu(xn @ Wg) * attn
        wmma_gemm_kernel<3><<<dim3(DIM / 64, ROWS / 64), 256, 0, stream>>>(
            xn, lWg, gbuf, nullptr, attn, ROWS, DIM, DIM);
        // x = gbuf @ Wo + x   (in-place residual: each element read once by its writer)
        wmma_gemm_kernel<5><<<dim3(DIM / 64, ROWS / 64), 256, 0, stream>>>(
            gbuf, lWo, x, nullptr, x, ROWS, DIM, DIM);

        // --- MLP block ---
        rmsnorm_kernel<<<ROWS, 256, 0, stream>>>(x, g_ff + (size_t)l * DIM, xn);
        wmma_gemm_kernel<2><<<dim3(FFDIM / 64, ROWS / 64), 256, 0, stream>>>(
            xn, lW1, ffh, lb1, nullptr, ROWS, FFDIM, DIM);
        wmma_gemm_kernel<4><<<dim3(DIM / 64, ROWS / 64), 256, 0, stream>>>(
            ffh, lW2, x, lb2, x, ROWS, DIM, FFDIM);
    }

    // --- final norm + vocab projection (dominates HBM traffic: 262 MB out) ---
    rmsnorm_kernel<<<ROWS, 256, 0, stream>>>(x, g_final, xn);
    wmma_gemm_kernel<0><<<dim3(VOCAB / 64, ROWS / 64), 256, 0, stream>>>(
        xn, Wl, out, nullptr, nullptr, ROWS, VOCAB, DIM);
}
